// Function_57303453663255
// MI455X (gfx1250) — compile-verified, matrix operand order checked by reading, not run on hardware
//
#include <hip/hip_runtime.h>

typedef _Float16 half_t;
typedef __attribute__((ext_vector_type(16))) _Float16 v16h;
typedef __attribute__((ext_vector_type(8)))  float    v8f;

#define BATCH   65536
#define IN      256
#define OUTCH   32
#define NCHEB   8
#define KTOT    2048      // IN * NCHEB
#define NCHUNK  64        // KTOT / 32 (WMMA K = 32)
#define WP_ELEMS (NCHUNK * 2 * 32 * 16)   // 65536 halves = 128 KB

// ---------------------------------------------------------------------------
// Pre-swizzle w [O=32][I=256][N=8] (f32) into f16 WMMA-B fragments.
// Fragment layout: wp[((chunk*2 + ntile)*32 + lane)*16 + h]
// For the 16-bit B operand of V_WMMA_F32_16X16X32_F16 (mirror of the
// documented A layout, M->N): lane<16 holds K={0..7,16..23}, lane>=16 holds
// K={8..15,24..31}; half h: v=h>>1 selects VGPR, h&1 selects lo/hi 16 bits.
// ---------------------------------------------------------------------------
__global__ void pack_w_kernel(const float* __restrict__ w, half_t* __restrict__ wp) {
    int idx = blockIdx.x * blockDim.x + threadIdx.x;
    if (idx >= WP_ELEMS) return;
    int h    = idx & 15;
    int lane = (idx >> 4) & 31;
    int frag = idx >> 9;          // 0..127
    int c    = frag >> 1;         // k-chunk 0..63
    int t    = frag & 1;          // n-tile 0..1
    int ncol = t * 16 + (lane & 15);
    int v  = h >> 1;
    int hi = h & 1;
    int K  = ((v < 4) ? (v * 2 + hi) : (16 + (v - 4) * 2 + hi)) + ((lane & 16) ? 8 : 0);
    int k  = c * 32 + K;          // flat k = i*8 + n
    wp[idx] = (half_t)w[ncol * KTOT + k];
}

// ---------------------------------------------------------------------------
// Main kernel: one wave -> 16 batch rows x all 32 outputs.
// A (Phi) fragments are generated in-register from x via the Chebyshev
// recurrence; B fragments come from LDS (whole 128 KB W staged per block,
// shared by all 8 waves). f32 accumulation via WMMA.
// ---------------------------------------------------------------------------
__global__ __launch_bounds__(256, 1)
void cheb_gemm_kernel(const float* __restrict__ x,
                      const half_t* __restrict__ wp,
                      float* __restrict__ out) {
    extern __shared__ __align__(16) half_t lds[];   // 65536 halves = 128 KB

    // Cooperative stage of all W fragments into LDS (16B per thread per iter).
    {
        const uint4* __restrict__ src = (const uint4*)wp;
        uint4* dst = (uint4*)lds;
        const int tid = threadIdx.x;
        #pragma unroll
        for (int it = 0; it < (WP_ELEMS * 2 / 16) / 256; ++it)   // 32 iters
            dst[tid + it * 256] = src[tid + it * 256];
    }
    __syncthreads();

    const int wave   = threadIdx.x >> 5;
    const int lane   = threadIdx.x & 31;
    const int tile   = blockIdx.x * 8 + wave;       // 4096 tiles of 16 rows
    const int row    = tile * 16 + (lane & 15);
    const int parity = (lane >> 4) & 1;             // which i within a pair

    const float4* __restrict__ xrow4 = (const float4*)(x + (size_t)row * IN);
    const v16h*   __restrict__ bfrag = (const v16h*)lds;

    v8f acc0 = {};
    v8f acc1 = {};

    #pragma unroll 4
    for (int c = 0; c < NCHUNK; ++c) {
        // Unconditional, branch-free prefetch ~8 chunks (128B) ahead; wrapped
        // index stays inside the same 1 KB row so it is always valid.
        __builtin_prefetch(&xrow4[(c + 8) & (NCHUNK - 1)], 0, 3);

        // x values this lane is responsible for in this 32-wide k-chunk
        const float4 xv = xrow4[c];                    // i0 = 4c .. 4c+3
        const float xa = parity ? xv.y : xv.x;         // i0 + parity
        const float xb = parity ? xv.w : xv.z;         // i0 + 2 + parity

        // Chebyshev T0..T7 (f32 recurrence, packed to f16 A-fragment):
        // halves 0..7  <- T_n(xa)  (K 0..7  / 8..15  by lane half)
        // halves 8..15 <- T_n(xb)  (K 16..23 / 24..31 by lane half)
        v16h a;
        float ta0 = 1.0f, ta1 = xa;
        float tb0 = 1.0f, tb1 = xb;
        a[0] = (half_t)ta0; a[1] = (half_t)ta1;
        a[8] = (half_t)tb0; a[9] = (half_t)tb1;
        #pragma unroll
        for (int n = 2; n < NCHEB; ++n) {
            float ta = 2.0f * xa * ta1 - ta0; ta0 = ta1; ta1 = ta;
            float tb = 2.0f * xb * tb1 - tb0; tb0 = tb1; tb1 = tb;
            a[n]     = (half_t)ta;
            a[8 + n] = (half_t)tb;
        }

        // B fragments from LDS: chunk c, n-tiles 0 and 1
        const v16h b0 = bfrag[c * 64 + lane];
        const v16h b1 = bfrag[c * 64 + lane + 32];

        acc0 = __builtin_amdgcn_wmma_f32_16x16x32_f16(
                   false, a, false, b0, (short)0, acc0, false, false);
        acc1 = __builtin_amdgcn_wmma_f32_16x16x32_f16(
                   false, a, false, b1, (short)0, acc1, false, false);
    }

    // D layout: lanes 0-15 -> M = r, lanes 16-31 -> M = r + 8; N = lane & 15
    const int mbase = tile * 16 + ((lane & 16) ? 8 : 0);
    const int ncol  = lane & 15;
    #pragma unroll
    for (int r = 0; r < 8; ++r) {
        out[(size_t)(mbase + r) * OUTCH + ncol]      = acc0[r];
        out[(size_t)(mbase + r) * OUTCH + 16 + ncol] = acc1[r];
    }
}

// ---------------------------------------------------------------------------
// Host entry
// ---------------------------------------------------------------------------
extern "C" void kernel_launch(void* const* d_in, const int* in_sizes, int n_in,
                              void* d_out, int out_size, void* d_ws, size_t ws_size,
                              hipStream_t stream) {
    const float* x = (const float*)d_in[0];   // [65536, 256] f32
    const float* w = (const float*)d_in[1];   // [32, 256, 8]  f32
    float* out     = (float*)d_out;           // [65536, 32]   f32
    half_t* wp     = (half_t*)d_ws;           // 128 KB fragment-packed W

    // Allow 128 KB dynamic LDS (WGP supports up to 320 KB per workgroup).
    (void)hipFuncSetAttribute(reinterpret_cast<const void*>(&cheb_gemm_kernel),
                              hipFuncAttributeMaxDynamicSharedMemorySize,
                              WP_ELEMS * (int)sizeof(half_t));

    pack_w_kernel<<<WP_ELEMS / 256, 256, 0, stream>>>(w, wp);
    cheb_gemm_kernel<<<BATCH / 16 / 8, 256, WP_ELEMS * sizeof(half_t), stream>>>(x, wp, out);
}
